// CXLoss_1030792151596
// MI455X (gfx1250) — matrix operand order repaired and checked by reading, not compiled, not run on hardware
//
#include <hip/hip_runtime.h>
#include <hip/hip_bf16.h>
#include <math.h>

typedef __attribute__((ext_vector_type(16))) _Float16 v16h;
typedef __attribute__((ext_vector_type(8)))  _Float16 v8h;
typedef __attribute__((ext_vector_type(8)))  float    v8f;

#define NN 4
#define CC 256
#define HH 48
#define WW 48
#define PP (HH*WW)          // 2304
#define SIGMA_INV 10.0f     // 1/sigma, sigma = 0.1
#define B_CONST 1.0f
#define EPS 1e-5f

// ---------------------------------------------------------------------------
// 1) Per-channel mean of featureT over (N,H,W). One block per channel.
// ---------------------------------------------------------------------------
__global__ __launch_bounds__(256) void mean_kernel(const float* __restrict__ T,
                                                   float* __restrict__ meanC) {
    int c   = blockIdx.x;
    int tid = threadIdx.x;
    const int total = NN * PP;
    float s = 0.f;
    for (int idx = tid; idx < total; idx += 256) {
        int n = idx / PP;
        int p = idx - n * PP;
        s += T[((size_t)(n * CC + c)) * PP + p];
    }
    __shared__ float sm[256];
    sm[tid] = s; __syncthreads();
    for (int st = 128; st > 0; st >>= 1) {
        if (tid < st) sm[tid] += sm[tid + st];
        __syncthreads();
    }
    if (tid == 0) meanC[c] = sm[0] / (float)total;
}

// ---------------------------------------------------------------------------
// 2) Center, L2-normalize over C, transpose to [n][p][C], split into f16
//    hi/lo pair (compensated precision). One block per (n,p), thread = channel.
// ---------------------------------------------------------------------------
__global__ __launch_bounds__(256) void pack_kernel(const float* __restrict__ T,
                                                   const float* __restrict__ I,
                                                   const float* __restrict__ meanC,
                                                   _Float16* __restrict__ Thi,
                                                   _Float16* __restrict__ Tlo,
                                                   _Float16* __restrict__ Ihi,
                                                   _Float16* __restrict__ Ilo) {
    int np = blockIdx.x;           // n*PP + p
    int n  = np / PP;
    int p  = np - n * PP;
    int c  = threadIdx.x;          // == channel, 256 threads
    float m = meanC[c];
    size_t idx = ((size_t)(n * CC + c)) * PP + p;
    float t  = T[idx] - m;
    float iv = I[idx] - m;
    __shared__ float sT[256], sI[256];
    sT[c] = t * t; sI[c] = iv * iv;
    __syncthreads();
    for (int s = 128; s > 0; s >>= 1) {
        if (c < s) { sT[c] += sT[c + s]; sI[c] += sI[c + s]; }
        __syncthreads();
    }
    float invT = 1.0f / sqrtf(sT[0]);
    float invI = 1.0f / sqrtf(sI[0]);
    float vt = t  * invT;
    float vi = iv * invI;
    size_t o = (size_t)np * CC + c;
    _Float16 th = (_Float16)vt;
    _Float16 ih = (_Float16)vi;
    Thi[o] = th;  Tlo[o] = (_Float16)(vt - (float)th);
    Ihi[o] = ih;  Ilo[o] = (_Float16)(vi - (float)ih);
}

// ---------------------------------------------------------------------------
// 3) Per-sample GEMM dist[p][q] = sum_c fT[p][c] * fI[q][c] via WMMA.
//    One wave per block; 32x32 output tile; split-f16 compensation:
//    dist ~= Ahi*Bhi + Ahi*Blo + Alo*Bhi   (f32 accumulate)
// ---------------------------------------------------------------------------
__device__ __forceinline__ v16h load_a_frag(const _Float16* __restrict__ base,
                                            int row0, int l16, int half, int k0) {
    // A 16x32 f16, contiguous k order used here (B matches): lanes 0-15 -> K k0..k0+15 of row row0+l16
    //                     lanes 16-31 -> K k0+16..k0+31
    const _Float16* ar = base + (size_t)(row0 + l16) * CC + k0 + half * 16;
    v8h lo = *(const v8h*)(ar);
    v8h hi = *(const v8h*)(ar + 8);
    return __builtin_shufflevector(lo, hi, 0,1,2,3,4,5,6,7,8,9,10,11,12,13,14,15);
}

__device__ __forceinline__ v16h load_b_frag(const _Float16* __restrict__ base,
                                            int col0, int l16, int half, int k0) {
    // B 32x16 f16, contiguous k order used here (A matches): lanes 0-15 -> K k0..k0+15 of column col0+l16
    //                     lanes 16-31 -> K k0+16..k0+31
    return *(const v16h*)(base + (size_t)(col0 + l16) * CC + k0 + half * 16);
}

__global__ __launch_bounds__(32) void gemm_kernel(const _Float16* __restrict__ Ahi,
                                                  const _Float16* __restrict__ Alo,
                                                  const _Float16* __restrict__ Bhi,
                                                  const _Float16* __restrict__ Blo,
                                                  float* __restrict__ dist) {
    int lane = threadIdx.x & 31;
    int half = lane >> 4;
    int l16  = lane & 15;
    int row0 = blockIdx.x * 32;
    int col0 = blockIdx.y * 32;

    v8f acc[2][2] = {};
    for (int k0 = 0; k0 < CC; k0 += 32) {
        v16h ah[2], al[2], bh[2], bl[2];
#pragma unroll
        for (int t = 0; t < 2; ++t) {
            ah[t] = load_a_frag(Ahi, row0 + t * 16, l16, half, k0);
            al[t] = load_a_frag(Alo, row0 + t * 16, l16, half, k0);
            bh[t] = load_b_frag(Bhi, col0 + t * 16, l16, half, k0);
            bl[t] = load_b_frag(Blo, col0 + t * 16, l16, half, k0);
        }
#pragma unroll
        for (int i = 0; i < 2; ++i) {
#pragma unroll
            for (int j = 0; j < 2; ++j) {
                acc[i][j] = __builtin_amdgcn_wmma_f32_16x16x32_f16(
                    false, ah[i], false, bh[j], (short)0, acc[i][j], false, false);
                acc[i][j] = __builtin_amdgcn_wmma_f32_16x16x32_f16(
                    false, ah[i], false, bl[j], (short)0, acc[i][j], false, false);
                acc[i][j] = __builtin_amdgcn_wmma_f32_16x16x32_f16(
                    false, al[i], false, bh[j], (short)0, acc[i][j], false, false);
            }
        }
    }
    // D layout: lanes 0-15: VGPR r -> (M = r, N = lane); lanes 16-31: (M = 8+r, N = lane-16)
#pragma unroll
    for (int i = 0; i < 2; ++i) {
        int r0 = row0 + i * 16 + half * 8;
#pragma unroll
        for (int j = 0; j < 2; ++j) {
            int cq = col0 + j * 16 + l16;
#pragma unroll
            for (int r = 0; r < 8; ++r)
                dist[(size_t)(r0 + r) * PP + cq] = acc[i][j][r];
        }
    }
}

// ---------------------------------------------------------------------------
// 4) Column pass: div[q] = min_p raw = (1 - max_p dist)/2, then
//    S[q] = sum_p exp((1 - raw/(div+eps))/sigma). One thread per column.
// ---------------------------------------------------------------------------
__global__ __launch_bounds__(256) void col_kernel(const float* __restrict__ dist,
                                                  float* __restrict__ divArr,
                                                  float* __restrict__ sArr) {
    int q = blockIdx.x * 256 + threadIdx.x;
    if (q >= PP) return;
    float mx = -3.402823466e+38f;
    for (int p = 0; p < PP; ++p)
        mx = fmaxf(mx, dist[(size_t)p * PP + q]);
    float dv  = (1.0f - mx) * 0.5f;
    float inv = 1.0f / (dv + EPS);
    float s = 0.f;
    for (int p = 0; p < PP; ++p) {
        float raw = (1.0f - dist[(size_t)p * PP + q]) * 0.5f;
        s += expf((B_CONST - raw * inv) * SIGMA_INV);
    }
    divArr[q] = dv;
    sArr[q]   = s;
}

// ---------------------------------------------------------------------------
// 5) Row pass: Mrow[p] = max_q  exp((1-raw/(div+eps))/sigma) / S[q].
//    One block per row (coalesced along the row).
// ---------------------------------------------------------------------------
__global__ __launch_bounds__(256) void row_kernel(const float* __restrict__ dist,
                                                  const float* __restrict__ divArr,
                                                  const float* __restrict__ sArr,
                                                  float* __restrict__ Mrow) {
    int p   = blockIdx.x;
    int tid = threadIdx.x;
    float mx = 0.f;   // CX >= 0
    for (int q = tid; q < PP; q += 256) {
        float raw = (1.0f - dist[(size_t)p * PP + q]) * 0.5f;
        float w   = expf((B_CONST - raw / (divArr[q] + EPS)) * SIGMA_INV);
        mx = fmaxf(mx, w / sArr[q]);
    }
    __shared__ float sm[256];
    sm[tid] = mx; __syncthreads();
    for (int s = 128; s > 0; s >>= 1) {
        if (tid < s) sm[tid] = fmaxf(sm[tid], sm[tid + s]);
        __syncthreads();
    }
    if (tid == 0) Mrow[p] = sm[0];
}

// ---------------------------------------------------------------------------
// 6) Per-sample finish: CX_B[n] = -log(mean_p Mrow[p])
// ---------------------------------------------------------------------------
__global__ __launch_bounds__(256) void finish_kernel(const float* __restrict__ Mrow,
                                                     float* __restrict__ out_n) {
    int tid = threadIdx.x;
    float s = 0.f;
    for (int p = tid; p < PP; p += 256) s += Mrow[p];
    __shared__ float sm[256];
    sm[tid] = s; __syncthreads();
    for (int st = 128; st > 0; st >>= 1) {
        if (tid < st) sm[tid] += sm[tid + st];
        __syncthreads();
    }
    if (tid == 0) *out_n = -logf(sm[0] / (float)PP);
}

// 7) loss = mean over samples
__global__ void loss_kernel(float* __restrict__ out) {
    if (threadIdx.x == 0)
        out[0] = 0.25f * (out[1] + out[2] + out[3] + out[4]);
}

// ---------------------------------------------------------------------------
extern "C" void kernel_launch(void* const* d_in, const int* in_sizes, int n_in,
                              void* d_out, int out_size, void* d_ws, size_t ws_size,
                              hipStream_t stream) {
    const float* T = (const float*)d_in[0];   // featureT [N,C,H,W] f32
    const float* I = (const float*)d_in[1];   // featureI [N,C,H,W] f32
    float* out = (float*)d_out;               // [loss, CX_B[0..3]]

    char* ws = (char*)d_ws;
    const size_t featHalfs = (size_t)NN * PP * CC;        // per split buffer
    float*    meanC = (float*)ws;                          // 1 KB
    _Float16* Thi   = (_Float16*)(ws + 1024);
    _Float16* Tlo   = Thi + featHalfs;
    _Float16* Ihi   = Tlo + featHalfs;
    _Float16* Ilo   = Ihi + featHalfs;
    float*    dist  = (float*)(ws + 1024 + 4 * featHalfs * sizeof(_Float16)); // 21.2 MB (per-sample reuse)
    float*    divA  = dist + (size_t)PP * PP;
    float*    sA    = divA + PP;
    float*    Mrow  = sA + PP;

    mean_kernel<<<CC, 256, 0, stream>>>(T, meanC);
    pack_kernel<<<NN * PP, 256, 0, stream>>>(T, I, meanC, Thi, Tlo, Ihi, Ilo);

    for (int n = 0; n < NN; ++n) {
        const _Float16* Ah = Thi + (size_t)n * PP * CC;
        const _Float16* Al = Tlo + (size_t)n * PP * CC;
        const _Float16* Bh = Ihi + (size_t)n * PP * CC;
        const _Float16* Bl = Ilo + (size_t)n * PP * CC;
        dim3 g(PP / 32, PP / 32);   // 72 x 72 waves
        gemm_kernel<<<g, 32, 0, stream>>>(Ah, Al, Bh, Bl, dist);
        col_kernel<<<(PP + 255) / 256, 256, 0, stream>>>(dist, divA, sA);
        row_kernel<<<PP, 256, 0, stream>>>(dist, divA, sA, Mrow);
        finish_kernel<<<1, 256, 0, stream>>>(Mrow, out + 1 + n);
    }
    loss_kernel<<<1, 32, 0, stream>>>(out);
}
